// ConditionalRBM_WithW_Stable_21165598834808
// MI455X (gfx1250) — matrix operand order repaired and checked by reading, not compiled, not run on hardware
//
#include <hip/hip_runtime.h>
#include <hip/hip_bf16.h>
#include <math.h>

#define V_DIM   256
#define H_DIM   256
#define BATCH   32768
#define K_STEPS 5
#define N_NOISE 3276           // int(32768 * 0.1)
#define GAMMA_W_MAX 0.05f

typedef _Float16 h16;
typedef __attribute__((ext_vector_type(16))) _Float16 v16h;
typedef __attribute__((ext_vector_type(8)))  _Float16 v8h;
typedef __attribute__((ext_vector_type(8)))  float    v8f;

// ---------------- utility device functions ----------------

__device__ __forceinline__ float urand(unsigned a, unsigned b, unsigned c){
  unsigned x = a * 0x9E3779B1u ^ (b + 0x7F4A7C15u) * 0x85EBCA77u ^ (c + 0x165667B1u) * 0xC2B2AE3Du;
  x ^= x >> 16; x *= 0x7FEB352Du;
  x ^= x >> 15; x *= 0x846CA68Bu;
  x ^= x >> 16;
  return (float)(x >> 8) * (1.0f / 16777216.0f);
}

// CDNA5 has a hardware V_TANH_F32 transcendental (TRANS32 class, co-executes
// with WMMA). Use it when the builtin exists; fall back to libm otherwise.
__device__ __forceinline__ float fast_tanhf(float x){
#if __has_builtin(__builtin_amdgcn_tanhf)
  return __builtin_amdgcn_tanhf(x);
#elif __has_builtin(__builtin_amdgcn_tanh_f32)
  return __builtin_amdgcn_tanh_f32(x);
#else
  return tanhf(x);
#endif
}

// sigmoid(x) = 0.5*tanh(x/2) + 0.5  -> single TRANS op + 2 VALU
__device__ __forceinline__ float sigmoidf(float x){
  return fmaf(0.5f, fast_tanhf(0.5f * x), 0.5f);
}

__device__ __forceinline__ float softplusf(float x){
  if (x >  20.0f) return x;
  if (x < -20.0f) return __expf(x);
  return log1pf(__expf(x));
}

__device__ __forceinline__ float neg_lse2(float fe_v, float fe_f){
  const float a = -fe_v, bq = -fe_f;
  const float m = fmaxf(a, bq);
  return -(m + logf(__expf(a - m) + __expf(bq - m)));
}

// ---------------- WMMA fragment helpers (CDNA5 wave32 layouts) ----------------
// A 16xK f16 tile from LDS (row-major, `stride` halves per row):
// lane L holds row M = L&15; halves j=0..7 -> K = kBase + (L>>4)*8 + j,
// halves j=8..15 -> K = kBase + (L>>4)*8 + 8 + j. Two contiguous 16B LDS reads.
__device__ __forceinline__ v16h load_A(const h16* base, int stride, int kBase, int lane){
  const int m = lane & 15, hi = lane >> 4;
  const h16* p = base + m * stride + kBase + hi * 8;
  v8h lo = *(const v8h*)p;
  v8h hv = *(const v8h*)(p + 16);
  v16h r;
  #pragma unroll
  for (int j = 0; j < 8; ++j){ r[j] = lo[j]; r[j + 8] = hv[j]; }
  return r;
}

// B 32x16 f16 from an LDS array laid out arr[n*stride + k] (element (k,n)):
// lane L holds column N = L&15; halves j=0..7 -> K = kBase + (L>>4)*8 + j,
// halves j=8..15 -> K = kBase + (L>>4)*8 + 8 + j. Two contiguous 16B reads.
__device__ __forceinline__ v16h load_B(const h16* base, int stride, int nBase, int kBase, int lane){
  const int n = lane & 15, hi = lane >> 4;
  const h16* p = base + (nBase + n) * stride + kBase + hi * 8;
  v8h lo = *(const v8h*)p;
  v8h hv = *(const v8h*)(p + 16);
  v16h r;
  #pragma unroll
  for (int j = 0; j < 8; ++j){ r[j] = lo[j]; r[j + 8] = hv[j]; }
  return r;
}

__device__ __forceinline__ v8f wmma32(v16h a, v16h b, v8f c){
  return __builtin_amdgcn_wmma_f32_16x16x32_f16(false, a, false, b, (short)0, c, false, false);
}

// ---------------- free energy for one 16-row tile held as A-fragments ----------------
__device__ void calc_fe(const v16h frag[8], int rowBase, int lane,
                        const h16* Wh, const float* wsum,
                        const float* b_mod, const float* c_mod, const float* w_scale,
                        float* scratch, float& fe_v, float& fe_f)
{
  float t2v[8], t2f[8];
  #pragma unroll
  for (int i = 0; i < 8; ++i){ t2v[i] = 0.f; t2f[i] = 0.f; }
  const int nl = lane & 15, hi = lane >> 4;

  #pragma unroll 1
  for (int tN = 0; tN < 16; ++tN){
    v8f acc = {0.f,0.f,0.f,0.f,0.f,0.f,0.f,0.f};
    #pragma unroll
    for (int kb = 0; kb < 8; ++kb)
      acc = wmma32(frag[kb], load_B(Wh, 256, tN * 16, kb * 32, lane), acc);
    const int n = tN * 16 + nl;
    const float Wn = wsum[n];
    #pragma unroll
    for (int i = 0; i < 8; ++i){
      const int g = rowBase + hi * 8 + i;              // C layout: M = hi*8 + i
      const float ws = w_scale[g * H_DIM + n];
      const float cm = c_mod[g * H_DIM + n];
      const float a  = acc[i];
      t2v[i] += softplusf(a * ws + cm);                // lin_v
      t2f[i] += softplusf((Wn - a) * ws + cm);         // lin_f
    }
  }
  // reduce over the 16 N-lanes of each half-wave
  #pragma unroll
  for (int i = 0; i < 8; ++i){
    #pragma unroll
    for (int mask = 1; mask < 16; mask <<= 1){
      t2v[i] += __shfl_xor(t2v[i], mask, 32);
      t2f[i] += __shfl_xor(t2f[i], mask, 32);
    }
  }
  // move per-row t2 totals to "row = lane&15" layout via wave-private LDS scratch
  if (nl == 0){
    #pragma unroll
    for (int i = 0; i < 8; ++i){
      scratch[hi * 8 + i]      = t2v[i];
      scratch[16 + hi * 8 + i] = t2f[i];
    }
  }
  // t1 terms from A-fragment layout: row = lane&15
  float s1 = 0.f, sB = 0.f;
  const int g = rowBase + nl;
  #pragma unroll
  for (int kb = 0; kb < 8; ++kb){
    #pragma unroll
    for (int j = 0; j < 16; ++j){
      const int k = kb * 32 + hi * 8 + j + ((j >= 8) ? 8 : 0);
      const float bm = b_mod[g * V_DIM + k];
      s1 += (float)frag[kb][j] * bm;
      sB += bm;
    }
  }
  s1 += __shfl_xor(s1, 16, 32);
  sB += __shfl_xor(sB, 16, 32);
  const float T2v = scratch[nl];
  const float T2f = scratch[16 + nl];
  fe_v = (-s1) - T2v;           // t1v - t2v
  fe_f = (s1 - sB) - T2f;       // t1f - t2f
}

// ---------------- conditional parameter network (WMMA) ----------------
// block = 128 threads (4 waves); wave owns 16 batch rows; grid = B/64.
// y = tanh(cond @ fc1_w.T + fc1_b) @ fc2_w.T + fc2_b, K = 64 both times.
// B(k,n) = fcX_w[n,k] -> fcX_w's row-major layout IS the B layout (stride 64).
// LDS: fc1h (8KB) | fc2h rows 0..1279 (160KB) | 4 x 2KB x-stage.
__global__ __launch_bounds__(128) void cond_kernel(
    const float* __restrict__ cond, const float* __restrict__ b, const float* __restrict__ c,
    const float* __restrict__ fc1_w, const float* __restrict__ fc1_b,
    const float* __restrict__ fc2_w, const float* __restrict__ fc2_b,
    float* __restrict__ b_mod, float* __restrict__ c_mod, float* __restrict__ w_scale)
{
  extern __shared__ char smem[];
  h16* fc1h   = (h16*)smem;                 // [64][64]
  h16* fc2h   = (h16*)(smem + 8192);        // [1280][64]
  h16* stageX = (h16*)(smem + 172032);      // 4 waves * 16*64

  const int tid  = threadIdx.x;
  const int lane = tid & 31;
  const int wave = tid >> 5;
  const int nl = lane & 15, hi = lane >> 4;

  for (int idx = tid; idx < 64 * 64; idx += 128) fc1h[idx] = (h16)fc1_w[idx];
  for (int idx = tid; idx < 1280 * 64; idx += 128){
    __builtin_prefetch(fc2_w + idx + 8 * 128, 0, 0);
    fc2h[idx] = (h16)fc2_w[idx];
  }
  __syncthreads();

  const int rowBase = blockIdx.x * 64 + wave * 16;
  h16* xstage = stageX + wave * (16 * 64);

  // cond A-fragments (16 rows x 64)
  v16h cfrag[2];
  {
    const int g = rowBase + nl;
    #pragma unroll
    for (int kb = 0; kb < 2; ++kb){
      v16h f;
      #pragma unroll
      for (int j = 0; j < 16; ++j){
        const int k = kb * 32 + hi * 8 + j + ((j >= 8) ? 8 : 0);
        f[j] = (h16)cond[g * 64 + k];
      }
      cfrag[kb] = f;
    }
  }

  // x = tanh(cond @ fc1_w.T + fc1_b) -> stage (A layout relayout via LDS)
  #pragma unroll
  for (int tN = 0; tN < 4; ++tN){
    v8f acc = {0.f,0.f,0.f,0.f,0.f,0.f,0.f,0.f};
    #pragma unroll
    for (int kb = 0; kb < 2; ++kb)
      acc = wmma32(cfrag[kb], load_B(fc1h, 64, tN * 16, kb * 32, lane), acc);
    const int o = tN * 16 + nl;
    const float bias = fc1_b[o];
    #pragma unroll
    for (int i = 0; i < 8; ++i)
      xstage[(hi * 8 + i) * 64 + o] = (h16)fast_tanhf(acc[i] + bias);
  }
  v16h xfrag[2];
  #pragma unroll
  for (int kb = 0; kb < 2; ++kb) xfrag[kb] = load_A(xstage, 64, kb * 32, lane);

  auto gemmTile = [&](int oBase) -> v8f {
    v8f acc = {0.f,0.f,0.f,0.f,0.f,0.f,0.f,0.f};
    #pragma unroll
    for (int kb = 0; kb < 2; ++kb)
      acc = wmma32(xfrag[kb], load_B(fc2h, 64, oBase, kb * 32, lane), acc);
    return acc;
  };

  // b_mod = (1 + gb) * b + bb
  #pragma unroll 1
  for (int t = 0; t < 16; ++t){
    v8f aG = gemmTile(t * 16);               // gb rows
    v8f aB = gemmTile(V_DIM + t * 16);       // bb rows
    const int i0 = t * 16 + nl;
    const float biasG = fc2_b[i0];
    const float biasB = fc2_b[V_DIM + i0];
    const float bval  = b[i0];
    #pragma unroll
    for (int i = 0; i < 8; ++i){
      const int g = rowBase + hi * 8 + i;
      b_mod[g * V_DIM + i0] = (1.0f + aG[i] + biasG) * bval + (aB[i] + biasB);
    }
  }
  // c_mod = (1 + gc) * c + bc
  #pragma unroll 1
  for (int t = 0; t < 16; ++t){
    v8f aG = gemmTile(2 * V_DIM + t * 16);           // gc
    v8f aB = gemmTile(2 * V_DIM + H_DIM + t * 16);   // bc
    const int i0 = t * 16 + nl;
    const float biasG = fc2_b[2 * V_DIM + i0];
    const float biasB = fc2_b[2 * V_DIM + H_DIM + i0];
    const float cval  = c[i0];
    #pragma unroll
    for (int i = 0; i < 8; ++i){
      const int g = rowBase + hi * 8 + i;
      c_mod[g * H_DIM + i0] = (1.0f + aG[i] + biasG) * cval + (aB[i] + biasB);
    }
  }
  // w_scale = 1 + 0.05 * tanh(gw)
  #pragma unroll 1
  for (int t = 0; t < 16; ++t){
    v8f aG = gemmTile(2 * V_DIM + 2 * H_DIM + t * 16);   // gw
    const int i0 = t * 16 + nl;
    const float biasG = fc2_b[2 * V_DIM + 2 * H_DIM + i0];
    #pragma unroll
    for (int i = 0; i < 8; ++i){
      const int g = rowBase + hi * 8 + i;
      w_scale[g * H_DIM + i0] = 1.0f + GAMMA_W_MAX * fast_tanhf(aG[i] + biasG);
    }
  }
}

// ---------------- fused Gibbs chain + free energy (WMMA) ----------------
// block = 128 threads (4 waves); wave owns 16 private batch rows; grid = B/64.
// LDS: Wh (W^T f16, 128KB) | Wv (W f16, 128KB) | wsum (1KB) | 4x8KB wave stage | fe scratch
__global__ __launch_bounds__(128) void rbm_gibbs_kernel(
    const float* __restrict__ v_data, const float* __restrict__ W,
    const float* __restrict__ b_mod, const float* __restrict__ c_mod,
    const float* __restrict__ w_scale, float* __restrict__ partials)
{
  extern __shared__ char smem[];
  h16*   Wh       = (h16*)  smem;                 // [H][V]: Wh[n*256+k] = W[k,n]
  h16*   Wv       = (h16*) (smem + 131072);       // [V][H]: = W
  float* wsum     = (float*)(smem + 262144);      // [H]
  h16*   stageAll = (h16*) (smem + 263168);       // 4 waves * 16*256 f16
  float* feScrAll = (float*)(smem + 295936);      // 4 waves * 32 f32

  const int tid  = threadIdx.x;
  const int lane = tid & 31;
  const int wave = tid >> 5;

  // cooperative W staging (both orientations, f32 -> f16)
  for (int idx = tid; idx < V_DIM * H_DIM; idx += 128){
    __builtin_prefetch(W + idx + 8 * 128, 0, 0);
    const float w = W[idx];
    const h16 wh = (h16)w;
    Wv[idx] = wh;
    Wh[(idx & 255) * 256 + (idx >> 8)] = wh;
  }
  for (int c0 = tid; c0 < H_DIM; c0 += 128){
    float s = 0.f;
    for (int r = 0; r < V_DIM; ++r) s += W[r * H_DIM + c0];
    wsum[c0] = s;
  }
  __syncthreads();

  const int rowBase = blockIdx.x * 64 + wave * 16;
  h16*   stage = stageAll + wave * (16 * 256);
  float* feScr = feScrAll + wave * 32;
  const int nl = lane & 15, hi = lane >> 4;

  // init v_model fragments: first N_NOISE rows replaced by Bernoulli(0.5) noise
  v16h vfrag[8];
  {
    const int g = rowBase + nl;
    #pragma unroll
    for (int kb = 0; kb < 8; ++kb){
      v16h f;
      #pragma unroll
      for (int j = 0; j < 16; ++j){
        const int k = kb * 32 + hi * 8 + j + ((j >= 8) ? 8 : 0);
        float val;
        if (g < N_NOISE) val = (urand((unsigned)g, (unsigned)k, 0x6E6F6973u) < 0.5f) ? 1.0f : 0.0f;
        else             val = v_data[g * V_DIM + k];
        f[j] = (h16)val;
      }
      vfrag[kb] = f;
    }
  }

  // ---- Gibbs chain: fully wave-local, no barriers ----
  #pragma unroll 1
  for (int step = 0; step < K_STEPS; ++step){
    // p_h = sigmoid((v@W)*w_scale + c_mod); sample h; stage hs = h*w_scale (f16)
    #pragma unroll 1
    for (int tN = 0; tN < 16; ++tN){
      v8f acc = {0.f,0.f,0.f,0.f,0.f,0.f,0.f,0.f};
      #pragma unroll
      for (int kb = 0; kb < 8; ++kb)
        acc = wmma32(vfrag[kb], load_B(Wh, 256, tN * 16, kb * 32, lane), acc);
      const int n = tN * 16 + nl;
      #pragma unroll
      for (int i = 0; i < 8; ++i){
        const int m = hi * 8 + i;
        const int g = rowBase + m;
        const float ws = w_scale[g * H_DIM + n];
        const float p  = sigmoidf(acc[i] * ws + c_mod[g * H_DIM + n]);
        const float hb = (urand((unsigned)g, (unsigned)(n | (step << 16)), 0x68626974u) < p) ? 1.0f : 0.0f;
        stage[m * 256 + n] = (h16)(hb * ws);
      }
    }
    v16h hfrag[8];
    #pragma unroll
    for (int kb = 0; kb < 8; ++kb) hfrag[kb] = load_A(stage, 256, kb * 32, lane);

    // p_v = sigmoid((h*w_scale)@W^T + b_mod); sample v'
    #pragma unroll 1
    for (int tN = 0; tN < 16; ++tN){
      v8f acc = {0.f,0.f,0.f,0.f,0.f,0.f,0.f,0.f};
      #pragma unroll
      for (int kb = 0; kb < 8; ++kb)
        acc = wmma32(hfrag[kb], load_B(Wv, 256, tN * 16, kb * 32, lane), acc);
      const int n = tN * 16 + nl;
      #pragma unroll
      for (int i = 0; i < 8; ++i){
        const int m = hi * 8 + i;
        const int g = rowBase + m;
        const float p  = sigmoidf(acc[i] + b_mod[g * V_DIM + n]);
        const float vb = (urand((unsigned)g, (unsigned)(n | (step << 16)), 0x76626974u) < p) ? 1.0f : 0.0f;
        stage[m * 256 + n] = (h16)vb;
      }
    }
    #pragma unroll
    for (int kb = 0; kb < 8; ++kb) vfrag[kb] = load_A(stage, 256, kb * 32, lane);
  }

  // ---- free energies ----
  float fev_m, fef_m;
  calc_fe(vfrag, rowBase, lane, Wh, wsum, b_mod, c_mod, w_scale, feScr, fev_m, fef_m);

  v16h dfrag[8];
  {
    const int g = rowBase + nl;
    #pragma unroll
    for (int kb = 0; kb < 8; ++kb){
      v16h f;
      #pragma unroll
      for (int j = 0; j < 16; ++j){
        const int k = kb * 32 + hi * 8 + j + ((j >= 8) ? 8 : 0);
        f[j] = (h16)v_data[g * V_DIM + k];
      }
      dfrag[kb] = f;
    }
  }
  float fev_d, fef_d;
  calc_fe(dfrag, rowBase, lane, Wh, wsum, b_mod, c_mod, w_scale, feScr, fev_d, fef_d);

  // per-row contribution: lanes 16..31 hold duplicates -> zero them
  float contrib = (hi == 0) ? (neg_lse2(fev_d, fef_d) - neg_lse2(fev_m, fef_m)) : 0.0f;
  #pragma unroll
  for (int mask = 1; mask < 32; mask <<= 1)
    contrib += __shfl_xor(contrib, mask, 32);
  if (lane == 0) partials[blockIdx.x * 4 + wave] = contrib;
}

// ---------------- deterministic final reduction ----------------
__global__ __launch_bounds__(256) void reduce_kernel(const float* __restrict__ partials,
                                                     int n, float* __restrict__ out, float scale)
{
  __shared__ float sh[256];
  float s = 0.f;
  for (int i = threadIdx.x; i < n; i += 256) s += partials[i];
  sh[threadIdx.x] = s;
  __syncthreads();
  for (int off = 128; off > 0; off >>= 1){
    if ((int)threadIdx.x < off) sh[threadIdx.x] += sh[threadIdx.x + off];
    __syncthreads();
  }
  if (threadIdx.x == 0) out[0] = sh[0] * scale;
}

// ---------------- launch ----------------
extern "C" void kernel_launch(void* const* d_in, const int* in_sizes, int n_in,
                              void* d_out, int out_size, void* d_ws, size_t ws_size,
                              hipStream_t stream)
{
  const float* v_data = (const float*)d_in[0];
  const float* cond   = (const float*)d_in[1];
  const float* W      = (const float*)d_in[2];
  const float* b      = (const float*)d_in[3];
  const float* c      = (const float*)d_in[4];
  const float* fc1_w  = (const float*)d_in[5];
  const float* fc1_b  = (const float*)d_in[6];
  const float* fc2_w  = (const float*)d_in[7];
  const float* fc2_b  = (const float*)d_in[8];
  float* out = (float*)d_out;

  char* ws = (char*)d_ws;
  const size_t matBytes = (size_t)BATCH * V_DIM * sizeof(float);   // 32 MB
  float* b_mod    = (float*)(ws);
  float* c_mod    = (float*)(ws + matBytes);
  float* w_scale  = (float*)(ws + 2 * matBytes);
  float* partials = (float*)(ws + 3 * matBytes);                   // 2048 floats

  const size_t condSmem = 8192u + 163840u + 8192u;                 // 180224
  cond_kernel<<<dim3(BATCH / 64), dim3(128), condSmem, stream>>>(
      cond, b, c, fc1_w, fc1_b, fc2_w, fc2_b, b_mod, c_mod, w_scale);

  const size_t gibbsSmem = 131072u * 2 + 1024u + 4u * 16u * 256u * 2u + 4u * 32u * 4u; // 296448
  rbm_gibbs_kernel<<<dim3(BATCH / 64), dim3(128), gibbsSmem, stream>>>(
      v_data, W, b_mod, c_mod, w_scale, partials);

  reduce_kernel<<<dim3(1), dim3(256), 0, stream>>>(partials, 2048, out, 1.0f / (float)BATCH);
}
